// SinkhornDistance_5987184411210
// MI455X (gfx1250) — matrix operand order repaired and checked by reading, not compiled, not run on hardware
//
#include <hip/hip_runtime.h>

typedef _Float16 v16h __attribute__((ext_vector_type(16)));
typedef _Float16 v8h  __attribute__((ext_vector_type(8)));
typedef float    v8f  __attribute__((ext_vector_type(8)));

#define N_DIM   1024
#define N_BATCH 64
#define MAX_IT  50
#define INV_EPS 10.0f   // 1/eps, eps = 0.1

// ---------------------------------------------------------------------------
// Kernel 1: K = exp(-C/eps) as f16, row-major, plus a transposed copy KT.
// grid (32, 32, 64), block (32, 8)
// ---------------------------------------------------------------------------
__global__ __launch_bounds__(256) void sinkhorn_prep(const float* __restrict__ C,
                                                     _Float16* __restrict__ K,
                                                     _Float16* __restrict__ KT) {
  __shared__ _Float16 tile[32][33];
  const int b  = blockIdx.z;
  const int i0 = blockIdx.y * 32;
  const int j0 = blockIdx.x * 32;
  const size_t boff = (size_t)b << 20;
  const int tx = threadIdx.x, ty = threadIdx.y;
#pragma unroll
  for (int r = ty; r < 32; r += 8) {
    const int i = i0 + r, j = j0 + tx;
    const float c = C[boff + (size_t)i * N_DIM + j];
    const _Float16 k = (_Float16)__expf(-INV_EPS * c);
    K[boff + (size_t)i * N_DIM + j] = k;
    tile[r][tx] = k;
  }
  __syncthreads();
#pragma unroll
  for (int r = ty; r < 32; r += 8) {
    KT[boff + (size_t)(j0 + r) * N_DIM + (i0 + tx)] = tile[tx][r];
  }
}

// ---------------------------------------------------------------------------
// One Sinkhorn half-step:  dst = num / (M * vec)   (M is 1024x1024 f16)
// vec is rescaled by s = 1/max(vec) before f16 conversion; s is divided
// back out in f32, so the f32 state never loses range.
// Each of the 32 waves computes two 16-row blocks with WMMA f16->f32.
// ---------------------------------------------------------------------------
__device__ __forceinline__ void sinkhorn_phase(const _Float16* __restrict__ M,
                                               const float* vec,
                                               const float* num,
                                               float* dst,
                                               _Float16* vh,
                                               unsigned* smax,
                                               const int tid) {
  if (tid == 0) *smax = 0u;
  __syncthreads();
  // positive floats compare like their bit patterns
  atomicMax(smax, __float_as_uint(vec[tid]));
  __syncthreads();
  const float s = 1.0f / __uint_as_float(*smax);
  vh[tid] = (_Float16)(vec[tid] * s);
  __syncthreads();

  const int lane = tid & 31;
  const int wave = tid >> 5;
#pragma unroll
  for (int blk = 0; blk < 2; ++blk) {
    const int i0 = (wave * 2 + blk) << 4;
    v8f acc = {};
    // A operand: rows of M, contiguous f16 A layout:
    // lanes 0-15 hold K 0..15, lanes 16-31 hold K 16..31
    const _Float16* rowp =
        M + (size_t)(i0 + (lane & 15)) * N_DIM + (lane & 16);
    // B operand: vec replicated across N; lanes 0-15 hold K 0..15,
    // lanes 16-31 hold K 16..31 (contiguous 32B from LDS)
    const _Float16* vbase = vh + (lane & 16);
#pragma unroll 4
    for (int j0 = 0; j0 < N_DIM; j0 += 32) {
      union { v16h v; v8h h[2]; } A;
      A.h[0] = *(const v8h*)(rowp + j0);        // K = base+0..7
      A.h[1] = *(const v8h*)(rowp + j0 + 8);    // K = base+8..15
      const v16h B = *(const v16h*)(vbase + j0);
      acc = __builtin_amdgcn_wmma_f32_16x16x32_f16(
          false, A.v, false, B, (short)0, acc, false, false);
    }
    // D layout: element (m,n): VGPR m&7, lane (m>>3)*16+n. Take column 0:
    // lane 0 holds rows i0+0..7, lane 16 holds rows i0+8..15.
    if ((lane & 15) == 0) {
      const int base = i0 + ((lane >> 4) << 3);
#pragma unroll
      for (int m = 0; m < 8; ++m)
        dst[base + m] = num[base + m] * s / acc[m];
    }
  }
  __syncthreads();
}

// ---------------------------------------------------------------------------
// Kernel 2: full 50-iteration Sinkhorn loop; one batch per workgroup.
// grid (64), block (1024) = 32 waves.
// ---------------------------------------------------------------------------
__global__ __launch_bounds__(1024, 1) void sinkhorn_iterate(
    const float* __restrict__ mu, const float* __restrict__ nu,
    const _Float16* __restrict__ K, const _Float16* __restrict__ KT,
    float* __restrict__ aOut, float* __restrict__ bOut) {
  const int b   = blockIdx.x;
  const int tid = threadIdx.x;

  __shared__ float sMu[N_DIM], sNu[N_DIM], sA[N_DIM], sB[N_DIM];
  __shared__ _Float16 __attribute__((aligned(32))) sVh[N_DIM];
  __shared__ unsigned sMax;

  sMu[tid] = mu[(b << 10) + tid];
  sNu[tid] = nu[(b << 10) + tid];
  sB[tid]  = 22026.465794806718f;  // b0 = exp(v0/eps) = e^10

  const _Float16* Kb  = K  + ((size_t)b << 20);
  const _Float16* KTb = KT + ((size_t)b << 20);

  for (int it = 0; it < MAX_IT; ++it) {
    sinkhorn_phase(Kb,  sB, sMu, sA, sVh, &sMax, tid);  // a = mu / (K b)
    sinkhorn_phase(KTb, sA, sNu, sB, sVh, &sMax, tid);  // b = nu / (K^T a)
  }

  aOut[(b << 10) + tid] = sA[tid];
  bOut[(b << 10) + tid] = sB[tid];
}

// ---------------------------------------------------------------------------
// Kernel 3: pi = a_i * exp(-C/eps) * b_j (full-precision C), partial costs.
// grid (256, 64), block (256)
// ---------------------------------------------------------------------------
__global__ __launch_bounds__(256) void sinkhorn_finalize(
    const float* __restrict__ C, const float* __restrict__ aV,
    const float* __restrict__ bV, float* __restrict__ pi,
    float* __restrict__ partials) {
  const int b = blockIdx.y;
  const float* Cb  = C  + ((size_t)b << 20);
  float*       pib = pi + ((size_t)b << 20);
  const float* ab  = aV + (b << 10);
  const float* bb  = bV + (b << 10);

  float sum = 0.0f;
  for (int idx = blockIdx.x * blockDim.x + threadIdx.x; idx < (1 << 20);
       idx += blockDim.x * gridDim.x) {
    const int i = idx >> 10, j = idx & (N_DIM - 1);
    const float c = Cb[idx];
    const float p = ab[i] * bb[j] * __expf(-INV_EPS * c);
    pib[idx] = p;
    sum += p * c;
  }
  const int lane = threadIdx.x & 31, wv = threadIdx.x >> 5;
#pragma unroll
  for (int off = 16; off > 0; off >>= 1) sum += __shfl_down(sum, off, 32);
  __shared__ float wsum[8];
  if (lane == 0) wsum[wv] = sum;
  __syncthreads();
  if (threadIdx.x == 0) {
    float t = 0.0f;
#pragma unroll
    for (int k = 0; k < 8; ++k) t += wsum[k];
    partials[(size_t)b * 256 + blockIdx.x] = t;
  }
}

// ---------------------------------------------------------------------------
// Kernel 4: deterministic cost reduction. grid (64), block (256)
// ---------------------------------------------------------------------------
__global__ __launch_bounds__(256) void sinkhorn_cost_reduce(
    const float* __restrict__ partials, float* __restrict__ cost) {
  const int b = blockIdx.x;
  float v = partials[(size_t)b * 256 + threadIdx.x];
  const int lane = threadIdx.x & 31, wv = threadIdx.x >> 5;
#pragma unroll
  for (int off = 16; off > 0; off >>= 1) v += __shfl_down(v, off, 32);
  __shared__ float wsum[8];
  if (lane == 0) wsum[wv] = v;
  __syncthreads();
  if (threadIdx.x == 0) {
    float t = 0.0f;
#pragma unroll
    for (int k = 0; k < 8; ++k) t += wsum[k];
    cost[b] = t;
  }
}

// ---------------------------------------------------------------------------
extern "C" void kernel_launch(void* const* d_in, const int* in_sizes, int n_in,
                              void* d_out, int out_size, void* d_ws,
                              size_t ws_size, hipStream_t stream) {
  (void)in_sizes; (void)n_in; (void)out_size; (void)ws_size;
  const float* mu = (const float*)d_in[0];   // [64,1024]
  const float* nu = (const float*)d_in[1];   // [64,1024]
  const float* C  = (const float*)d_in[2];   // [64,1024,1024]

  float* out  = (float*)d_out;
  float* cost = out;        // [64]
  float* pi   = out + 64;   // [64,1024,1024]

  char* ws = (char*)d_ws;
  const size_t kBytes = (size_t)N_BATCH * N_DIM * N_DIM * sizeof(_Float16);
  _Float16* K  = (_Float16*)ws;
  _Float16* KT = (_Float16*)(ws + kBytes);
  float* aV       = (float*)(ws + 2 * kBytes);
  float* bV       = aV + N_BATCH * N_DIM;
  float* partials = bV + N_BATCH * N_DIM;   // [64,256]

  sinkhorn_prep<<<dim3(32, 32, 64), dim3(32, 8), 0, stream>>>(C, K, KT);
  sinkhorn_iterate<<<dim3(64), dim3(1024), 0, stream>>>(mu, nu, K, KT, aV, bV);
  sinkhorn_finalize<<<dim3(256, 64), dim3(256), 0, stream>>>(C, aV, bV, pi,
                                                             partials);
  sinkhorn_cost_reduce<<<dim3(64), dim3(256), 0, stream>>>(partials, cost);
}
